// g_a_38972533243994
// MI455X (gfx1250) — 8-bit matrix operand order repaired and checked by reading, compiled, not run on hardware
//
#include <hip/hip_runtime.h>
#include <stdint.h>
#include <stddef.h>

typedef int  v8i __attribute__((ext_vector_type(8)));
typedef int  v4i __attribute__((ext_vector_type(4)));
typedef int  v2i __attribute__((ext_vector_type(2)));
typedef unsigned int v4u __attribute__((ext_vector_type(4)));

#if defined(__has_builtin)
#if __has_builtin(__builtin_amdgcn_tensor_load_to_lds) && \
    __has_builtin(__builtin_amdgcn_s_wait_tensorcnt)
#define USE_TDM 1
#endif
#endif

#define WG 256


__global__ void quant_input_kernel(const float* __restrict__ x,
                                   uint8_t* __restrict__ out,
                                   int NI, int C, int H, int W) {
  int i = blockIdx.x * blockDim.x + threadIdx.x;
  int total = NI * C * H * W;
  if (i >= total) return;
  int c = i % C;
  int t = i / C;
  int w = t % W;
  t /= W;
  int h = t % H;
  int n = t / H;
  float v = rintf(x[(((size_t)n * C + c) * H + h) * W + w] * 256.0f);
  v = fminf(fmaxf(v, 0.0f), 255.0f);
  out[i] = (uint8_t)v;
}

__global__ void quant_weight_kernel(const float* __restrict__ w,
                                    int8_t* __restrict__ wq,
                                    int OC, int Ci, int Kpad) {
  int i = blockIdx.x * blockDim.x + threadIdx.x;
  if (i >= OC * Kpad) return;
  int oc = i / Kpad, k = i - oc * Kpad;
  int8_t q = 0;
  int K = 25 * Ci;
  if (k < K) {
    int tap = k / Ci;
    int cin = k - tap * Ci;
    q = (int8_t)(int)rintf(w[((size_t)oc * Ci + cin) * 25 + tap]);
  }
  wq[i] = q;
}

__global__ void quant_bias_kernel(const float* __restrict__ b,
                                  int* __restrict__ bi, int n, float scale) {
  int i = blockIdx.x * blockDim.x + threadIdx.x;
  if (i < n) bi[i] = (int)rintf(b[i] * scale);
}


#if USE_TDM
__device__ __forceinline__ void tdm_load_tile(
    uint32_t lds_off, const void* gsrc, uint32_t rows, uint32_t rowBytes,
    uint32_t strideBytes, uint32_t tdim0, uint32_t tdim1) {
  uint64_t ga = (uint64_t)(uintptr_t)gsrc;
  v4u g0 = {1u, lds_off, (uint32_t)ga,
            (uint32_t)((ga >> 32) & 0x1FFFFFFu) | (2u << 30)};
  v8i g1;
  g1[0] = 0;
  g1[1] = (int)((tdim0 & 0xFFFFu) << 16);
  g1[2] = (int)(((tdim0 >> 16) & 0xFFFFu) | ((tdim1 & 0xFFFFu) << 16));
  g1[3] = (int)(((tdim1 >> 16) & 0xFFFFu) | ((rowBytes & 0xFFFFu) << 16));
  g1[4] = (int)(rows & 0xFFFFu);
  g1[5] = (int)strideBytes;
  g1[6] = 0;
  g1[7] = 0;
  v4i gz = {0, 0, 0, 0};
#if __clang_major__ >= 23
  v8i gz8 = {0, 0, 0, 0, 0, 0, 0, 0};
  __builtin_amdgcn_tensor_load_to_lds(g0, g1, gz, gz, gz8, 0);
#else
  __builtin_amdgcn_tensor_load_to_lds(g0, g1, gz, gz, 0);
#endif
}
#endif


struct Smem {
  uint8_t A[2][256][64];
  int8_t  B[2][64][64];
};

template <int MODE>
__global__ __launch_bounds__(WG) void conv_wmma_kernel(
    const uint8_t* __restrict__ actIn, const int8_t* __restrict__ wq,
    const int* __restrict__ bias, const float* __restrict__ mul,
    uint8_t* __restrict__ actOut, float* __restrict__ fOut,
    int Ci, int Hi, int Wi, int WoShift, int OC, int Kpad,
    const int* __restrict__ mdPtr, const int* __restrict__ reluPtr,
    const int* __restrict__ gaPtr) {

  __shared__ Smem smem;

  const int tid  = threadIdx.x;
  const int lane = tid & 31;
  const int wave = tid >> 5;
  const int n       = blockIdx.z;
  const int pixBase = blockIdx.x * 256;
  const int ocBase  = blockIdx.y * 64;
  const int half = lane >> 4;
  const int mrow = lane & 15;
  const int Wo = 1 << WoShift, WoMask = Wo - 1, Ho = Hi >> 1;
  const int K  = 25 * Ci;

  auto fillB = [&](int buf, int k0) {
#if USE_TDM
    if (wave == 0)
      tdm_load_tile((uint32_t)(offsetof(Smem, B) + (uint32_t)buf * 4096),
                    wq + (size_t)ocBase * Kpad + k0,
                    64, 64, (uint32_t)Kpad, (uint32_t)Kpad, (uint32_t)OC);
#else
    const int row = tid >> 2, q = tid & 3;
    *(uint4*)&smem.B[buf][row][q * 16] =
        *(const uint4*)(wq + (size_t)(ocBase + row) * Kpad + k0 + q * 16);
#endif
  };

  auto fillA = [&](int buf, int k0) {
    if (MODE != 0) {
      const int tap  = k0 / Ci;
      const int cin0 = k0 - tap * Ci;
      const int ky = tap / 5, kx = tap - 5 * ky;
      const int row = tid;
      const int p  = pixBase + row;
      const int oy = p >> WoShift, ox = p & WoMask;
      const int iy = 2 * oy - 2 + ky;
      const int ix = 2 * ox - 2 + kx;
      uint4 d[4] = {};
      if (iy >= 0 && iy < Hi && ix >= 0 && ix < Wi) {
        const uint4* src = (const uint4*)(actIn +
            (((size_t)n * Hi + iy) * Wi + ix) * Ci + cin0);
#pragma unroll
        for (int j = 0; j < 4; ++j) d[j] = src[j];
      }
      uint4* dst = (uint4*)&smem.A[buf][row][0];
#pragma unroll
      for (int j = 0; j < 4; ++j) dst[j] = d[j];
    } else {
#pragma unroll 1
      for (int j = 0; j < 64; ++j) {
        int e   = tid + j * 256;
        int row = e >> 6;
        int kk  = e & 63;
        int k   = k0 + kk;
        uint8_t val = 0;
        if (k < K) {
          int tap = k / Ci;
          int cin = k - tap * Ci;
          int ky = tap / 5, kx = tap - 5 * ky;
          int p  = pixBase + row;
          int oy = p >> WoShift, ox = p & WoMask;
          int iy = 2 * oy - 2 + ky;
          int ix = 2 * ox - 2 + kx;
          if (iy >= 0 && iy < Hi && ix >= 0 && ix < Wi)
            val = actIn[(((size_t)n * Hi + iy) * Wi + ix) * Ci + cin];
        }
        smem.A[buf][row][kk] = val;
      }
    }
  };

  v8i zero8 = 0;
  v8i acc[2][4];
#pragma unroll
  for (int t = 0; t < 2; ++t)
#pragma unroll
    for (int u = 0; u < 4; ++u) acc[t][u] = zero8;

  const int nChunks = Kpad >> 6;
  fillA(0, 0);
  fillB(0, 0);
  for (int kc = 0; kc < nChunks; ++kc) {
    const int cur = kc & 1;
#if USE_TDM
    if (wave == 0) __builtin_amdgcn_s_wait_tensorcnt(0);
#endif
    __syncthreads();
    if (kc + 1 < nChunks) {
      fillA(1 - cur, (kc + 1) << 6);
      fillB(1 - cur, (kc + 1) << 6);
    }

    v8i afrag[2];
#pragma unroll
    for (int t = 0; t < 2; ++t) {
      const uint8_t* ap = &smem.A[cur][wave * 32 + t * 16 + mrow][0];
#pragma unroll
      for (int j = 0; j < 4; ++j) {
        v2i tt = *(const v2i*)(ap + half * 16 + (j >> 1) * 32 + (j & 1) * 8);
        afrag[t][2 * j]     = tt.x;
        afrag[t][2 * j + 1] = tt.y;
      }
    }
#pragma unroll
    for (int u = 0; u < 4; ++u) {
      const int8_t* bp = &smem.B[cur][u * 16 + mrow][0];
      v4i t0 = *(const v4i*)(bp + half * 16);
      v4i t1 = *(const v4i*)(bp + 32 + half * 16);
      v8i bfrag;
#pragma unroll
      for (int j = 0; j < 4; ++j) {
        bfrag[j]     = t0[j];
        bfrag[4 + j] = t1[j];
      }
      acc[0][u] = __builtin_amdgcn_wmma_i32_16x16x64_iu8(false, afrag[0], true,
                                                         bfrag, acc[0][u], false, false);
      acc[1][u] = __builtin_amdgcn_wmma_i32_16x16x64_iu8(false, afrag[1], true,
                                                         bfrag, acc[1][u], false, false);
    }
  }

  const int md   = *mdPtr;
  const int relu = *reluPtr;
  const int ga   = *gaPtr;
  const int shift1 = (MODE == 0) ? md : (MODE == 1 ? md - 8 : md - ga);
  long long clp = 0, scl = 0;
  if (MODE != 2) {
    clp = llrint(255.0 * 16777216.0 / (double)relu);
    scl = (long long)((relu + 4) >> 3);
  }
  const long long rnd1 = 1LL << (shift1 - 1);

#pragma unroll
  for (int t = 0; t < 2; ++t) {
#pragma unroll
    for (int u = 0; u < 4; ++u) {
      const int oc = ocBase + u * 16 + mrow;
      const long long m  = llrintf(mul[oc]);
      const long long bv = bias[oc];
      v8i a = acc[t][u];
#pragma unroll
      for (int r = 0; r < 8; ++r) {
        int p  = pixBase + wave * 32 + t * 16 + r + half * 8;
        int oy = p >> WoShift, ox = p & WoMask;
        long long v = ((long long)a[r] + bv) * m;
        v = (v + rnd1) >> shift1;
        if (MODE == 2) {
          fOut[(((size_t)n * OC + oc) * Ho + oy) * Wo + ox] = (float)v;
        } else {
          if (v < 0)   v = 0;
          if (v > clp) v = clp;
          v = (v * scl + (1LL << 20)) >> 21;
          actOut[(((size_t)n * Ho + oy) * Wo + ox) * OC + oc] = (uint8_t)v;
        }
      }
    }
  }
}


extern "C" void kernel_launch(void* const* d_in, const int* in_sizes, int n_in,
                              void* d_out, int out_size, void* d_ws,
                              size_t ws_size, hipStream_t stream) {
  const float* x    = (const float*)d_in[0];
  const float* w0   = (const float*)d_in[1];
  const float* b0   = (const float*)d_in[2];
  const float* w1   = (const float*)d_in[3];
  const float* b1   = (const float*)d_in[4];
  const float* w2   = (const float*)d_in[5];
  const float* b2   = (const float*)d_in[6];
  const float* w3   = (const float*)d_in[7];
  const float* b3   = (const float*)d_in[8];
  const float* mul0 = (const float*)d_in[9];
  const float* mul1 = (const float*)d_in[10];
  const float* mul2 = (const float*)d_in[11];
  const float* mul3 = (const float*)d_in[12];
  const int* relu0  = (const int*)d_in[13];
  const int* relu1  = (const int*)d_in[14];
  const int* relu2  = (const int*)d_in[15];
  const int* md0    = (const int*)d_in[16];
  const int* md1    = (const int*)d_in[17];
  const int* md2    = (const int*)d_in[18];
  const int* md3    = (const int*)d_in[19];
  const int* gal    = (const int*)d_in[20];

  const int NI = 4, N = 192, M = 320;

  uint8_t* ws = (uint8_t*)d_ws;
  size_t off = 0;
  auto alloc = [&](size_t bytes) -> void* {
    void* p = ws + off;
    off = (off + bytes + 255) & ~(size_t)255;
    return p;
  };

  uint8_t* act0 = (uint8_t*)alloc((size_t)NI * 3 * 512 * 512);
  uint8_t* act1 = (uint8_t*)alloc((size_t)NI * N * 256 * 256);
  uint8_t* act2 = (uint8_t*)alloc((size_t)NI * N * 128 * 128);
  uint8_t* act3 = (uint8_t*)alloc((size_t)NI * N * 64 * 64);
  int8_t*  wq0  = (int8_t*)alloc((size_t)N * 128);
  int8_t*  wq1  = (int8_t*)alloc((size_t)N * 4800);
  int8_t*  wq2  = (int8_t*)alloc((size_t)N * 4800);
  int8_t*  wq3  = (int8_t*)alloc((size_t)M * 4800);
  int* bi0 = (int*)alloc(N * sizeof(int));
  int* bi1 = (int*)alloc(N * sizeof(int));
  int* bi2 = (int*)alloc(N * sizeof(int));
  int* bi3 = (int*)alloc(M * sizeof(int));

  {
    int total = NI * 3 * 512 * 512;
    quant_input_kernel<<<(total + 255) / 256, 256, 0, stream>>>(x, act0, NI, 3, 512, 512);
  }
  quant_weight_kernel<<<(N * 128 + 255) / 256, 256, 0, stream>>>(w0, wq0, N, 3, 128);
  quant_weight_kernel<<<(N * 4800 + 255) / 256, 256, 0, stream>>>(w1, wq1, N, N, 4800);
  quant_weight_kernel<<<(N * 4800 + 255) / 256, 256, 0, stream>>>(w2, wq2, N, N, 4800);
  quant_weight_kernel<<<(M * 4800 + 255) / 256, 256, 0, stream>>>(w3, wq3, M, N, 4800);
  quant_bias_kernel<<<1, 256, 0, stream>>>(b0, bi0, N, 256.0f);
  quant_bias_kernel<<<1, 256, 0, stream>>>(b1, bi1, N, 1.0f);
  quant_bias_kernel<<<1, 256, 0, stream>>>(b2, bi2, N, 1.0f);
  quant_bias_kernel<<<2, 256, 0, stream>>>(b3, bi3, M, 1.0f);

  conv_wmma_kernel<0><<<dim3(256 * 256 / 256, N / 64, NI), WG, 0, stream>>>(
      act0, wq0, bi0, mul0, act1, (float*)d_out,
      3, 512, 512,  8, N, 128, md0, relu0, gal);
  conv_wmma_kernel<1><<<dim3(128 * 128 / 256, N / 64, NI), WG, 0, stream>>>(
      act1, wq1, bi1, mul1, act2, (float*)d_out,
      N, 256, 256,  7, N, 4800, md1, relu1, gal);
  conv_wmma_kernel<1><<<dim3(64 * 64 / 256, N / 64, NI), WG, 0, stream>>>(
      act2, wq2, bi2, mul2, act3, (float*)d_out,
      N, 128, 128,  6, N, 4800, md2, relu2, gal);
  conv_wmma_kernel<2><<<dim3(32 * 32 / 256, M / 64, NI), WG, 0, stream>>>(
      act3, wq3, bi3, mul3, act3  , (float*)d_out,
      N, 64, 64,  5, M, 4800, md3, relu0, gal);
}
